// Deep_Lagrangian_Network_73589969649893
// MI455X (gfx1250) — matrix operand order repaired and checked by reading, not compiled, not run on hardware
//
#include <hip/hip_runtime.h>
#include <hip/hip_bf16.h>

typedef __attribute__((ext_vector_type(16))) _Float16 v16h;
typedef __attribute__((ext_vector_type(8)))  _Float16 v8h;
typedef __attribute__((ext_vector_type(8)))  float    v8f;

#define HIDN 512
#define NDOF 7
#define NTRIL 21
#define LDP  520   // padded LDS stride in halves (1040B -> bank offset 4*lane, conflict-free)
#define HLD  520   // padded activation stride in halves

// LDS layout (dynamic, 283648 bytes total):
//   [0, 133120)        Tin  : 128 cols x LDP halves (8 pairs x 16 tangent cols)
//   [133120, 266240)   Tout : 128 cols x LDP halves  (forward hA/hB alias here)
//   [266240, 283648)   masks mk0/1/2, diag_s, tril_s, Jd, Jt
#define SMEM_TIN   0
#define SMEM_TOUT  133120
#define SMEM_MISC  266240
#define SMEM_BYTES 283648

// ---------------- WMMA helpers (CDNA5 gfx1250, wave32) ----------------

// D = A(16x32 f16) x B(32x16 f16) + C(16x16 f32). Reuse flags must be ICEs.
__device__ __forceinline__ v8f wmma_f16(v16h a, v16h b, v8f c) {
  return __builtin_amdgcn_wmma_f32_16x16x32_f16(false, a, false, b, (short)0, c,
                                                false, false);
}
// VOP3P OPSEL RA hint: this WMMA's A operand (same VGPRs) is identical to the
// previous WMMA's A operand in an unrolled chain.
__device__ __forceinline__ v8f wmma_f16_ra(v16h a, v16h b, v8f c) {
  return __builtin_amdgcn_wmma_f32_16x16x32_f16(false, a, false, b, (short)0, c,
                                                true, false);
}

// A-matrix fragment: row-major f16 source, ld = row stride in halves.
// ISA 16-bit A 16x32: lanes0-15 halves {K0-7,K16-23}; lanes16-31 {K8-15,K24-31}.
__device__ __forceinline__ v16h ldA(const _Float16* base, int ld, int lane) {
  const int r  = lane & 15;
  const int hi = (lane >> 4) << 3;
  const _Float16* p = base + (size_t)r * ld + hi;
  v8h x = *(const v8h*)(p);
  v8h y = *(const v8h*)(p + 16);
  v16h a;
#pragma unroll
  for (int i = 0; i < 8; ++i) { a[i] = x[i]; a[i + 8] = y[i]; }
  return a;
}

// B-matrix fragment: each of the 16 B-columns stored k-contiguously with stride ld.
// ISA 16-bit B 32x16: lanes0-15 hold {K0-7,K16-23}, lanes16-31 {K8-15,K24-31}.
__device__ __forceinline__ v16h ldBv(const _Float16* base, int ld, int lane) {
  const int c  = lane & 15;
  const int ko = (lane >> 4) << 3;
  const _Float16* p = base + (size_t)c * ld + ko;
  v8h x = *(const v8h*)(p);
  v8h y = *(const v8h*)(p + 16);
  v16h b;
#pragma unroll
  for (int i = 0; i < 8; ++i) { b[i] = x[i]; b[i + 8] = y[i]; }
  return b;
}

// ---------------- prep kernels (f32 -> f16 weight staging) ----------------

__global__ void k_prep_q(const float* __restrict__ q, _Float16* __restrict__ qh, int B) {
  int idx = blockIdx.x * 256 + threadIdx.x;
  if (idx >= B * 32) return;
  int r = idx >> 5, c = idx & 31;
  qh[idx] = (c < NDOF) ? (_Float16)q[r * NDOF + c] : (_Float16)0.f;
}

__global__ void k_prep_w0(const float* __restrict__ W0, _Float16* __restrict__ W0h) {
  int idx = blockIdx.x * 256 + threadIdx.x;
  if (idx >= HIDN * 32) return;
  int k = idx >> 5, j = idx & 31;
  W0h[idx] = (j < NDOF) ? (_Float16)W0[k * NDOF + j] : (_Float16)0.f;
}

// column-major f16 copy of W0: W0c[j][k] = W0[k][j], 8 rows (row 7 zero)
__global__ void k_prep_w0c(const float* __restrict__ W0, _Float16* __restrict__ W0c) {
  int idx = blockIdx.x * 256 + threadIdx.x;
  if (idx >= 8 * HIDN) return;
  int j = idx >> 9, k = idx & 511;
  W0c[idx] = (j < NDOF) ? (_Float16)W0[k * NDOF + j] : (_Float16)0.f;
}

__global__ void k_cvt(const float* __restrict__ src, _Float16* __restrict__ dst, int n) {
  int idx = blockIdx.x * 256 + threadIdx.x;
  if (idx < n) dst[idx] = (_Float16)src[idx];
}

__global__ void k_prep_heads(const float* __restrict__ Wd, const float* __restrict__ Wt,
                             _Float16* __restrict__ WH) {
  int idx = blockIdx.x * 256 + threadIdx.x;
  if (idx >= 32 * HIDN) return;
  int r = idx >> 9, k = idx & 511;
  float v = 0.f;
  if (r < NDOF) v = Wd[r * HIDN + k];
  else if (r < NDOF + NTRIL) v = Wt[(r - NDOF) * HIDN + k];
  WH[idx] = (_Float16)v;
}

// ---------------- device helpers ----------------

// forward layer: M=16 samples (A in LDS), N=512 (4 n-tiles per wave), K=512
__device__ __forceinline__ void fwd_layer(const _Float16* hin, const _Float16* Wh,
                                          const float* __restrict__ bias,
                                          _Float16* hout, unsigned* mk,
                                          int lane, int wave) {
  const int colL = lane & 15;
  const int rowB = (lane >> 4) << 3;
  v8f z = {0, 0, 0, 0, 0, 0, 0, 0};
  v8f acc[4] = {z, z, z, z};
  for (int k0 = 0; k0 < HIDN; k0 += 32) {
    v16h a = ldA(hin + k0, HLD, lane);
    {  // u = 0 (no reuse: fresh A)
      v16h b = ldBv(Wh + ((size_t)(wave * 4 + 0) * 16) * HIDN + k0, HIDN, lane);
      acc[0] = wmma_f16(a, b, acc[0]);
    }
#pragma unroll
    for (int u = 1; u < 4; ++u) {   // same A across the 4-chain
      v16h b = ldBv(Wh + ((size_t)(wave * 4 + u) * 16) * HIDN + k0, HIDN, lane);
      acc[u] = wmma_f16_ra(a, b, acc[u]);
    }
  }
#pragma unroll
  for (int u = 0; u < 4; ++u) {
    const int n = (wave * 4 + u) * 16 + colL;
    const float bv = bias[n];
#pragma unroll
    for (int j = 0; j < 8; ++j) {
      const int sl = rowB + j;
      float pre = acc[u][j] + bv;
      hout[sl * HLD + n] = (_Float16)(pre > 0.f ? pre : 0.f);
      // branchless: ds_or of 0 when inactive (no exec-mask dance)
      atomicOr(&mk[sl * 16 + (n >> 5)], pre > 0.f ? (1u << (n & 31)) : 0u);
    }
  }
}

// batched Jacobian level: D(512x128) = W(512x512) @ Tin(512x128); weight A-tile
// is loaded ONCE per (mt,k0) and reused by 8 WMMAs (one per sample pair).
__device__ __forceinline__ void jac_level(const _Float16* Wh, const _Float16* Tin,
                                          const unsigned* mk, _Float16* Tout,
                                          int lane, int wave) {
  const int c  = lane & 15;
  const int rb = (lane >> 4) << 3;
  v8f z = {0, 0, 0, 0, 0, 0, 0, 0};
  for (int mi = 0; mi < 4; ++mi) {
    const int mt = wave * 4 + mi;
    v8f acc[8] = {z, z, z, z, z, z, z, z};
    for (int k0 = 0; k0 < HIDN; k0 += 32) {
      v16h a = ldA(Wh + ((size_t)mt * 16) * HIDN + k0, HIDN, lane);
      {  // g = 0 (no reuse: fresh A)
        v16h b = ldBv(Tin + (size_t)0 * LDP + k0, LDP, lane);
        acc[0] = wmma_f16(a, b, acc[0]);
      }
#pragma unroll
      for (int g = 1; g < 8; ++g) {   // same A across the 8-chain
        v16h b = ldBv(Tin + (size_t)(g * 16) * LDP + k0, LDP, lane);
        acc[g] = wmma_f16_ra(a, b, acc[g]);
      }
    }
    const int krow = mt * 16 + rb;
#pragma unroll
    for (int g = 0; g < 8; ++g) {
      v8h tv;
      if (c < 14) {
        const int s = g * 2 + ((c < 7) ? 0 : 1);
        const unsigned mw = mk[s * 16 + (krow >> 5)];
        const int sh = krow & 31;
#pragma unroll
        for (int j = 0; j < 8; ++j)
          tv[j] = (_Float16)(((mw >> (sh + j)) & 1u) ? acc[g][j] : 0.f);
      } else {
#pragma unroll
        for (int j = 0; j < 8; ++j) tv[j] = (_Float16)0.f;
      }
      *(v8h*)(Tout + (size_t)(g * 16 + c) * LDP + krow) = tv;
    }
  }
}

__device__ __forceinline__ float Lget(const float* dg, const float* tl, int g, int i, int j) {
  if (i == j) return dg[g * 7 + i] + 0.1f;
  if (i > j)  return tl[g * 21 + (i * (i - 1)) / 2 + j];
  return 0.f;
}
__device__ __forceinline__ float LdqGet(const float* Jd, const float* Jt, int g,
                                        int i, int j, int k) {
  if (i == j) return Jd[g * 49 + i * 7 + k];
  if (i > j)  return Jt[g * 147 + ((i * (i - 1)) / 2 + j) * 7 + k];
  return 0.f;
}

// ---------------- fused mega kernel: 16 samples per workgroup ----------------

__global__ __launch_bounds__(256)
void delan_mega(const float* __restrict__ qd,
                const float* __restrict__ b0, const float* __restrict__ b1,
                const float* __restrict__ b2, const float* __restrict__ bd,
                const float* __restrict__ bt,
                const _Float16* __restrict__ qh, const _Float16* __restrict__ W0h,
                const _Float16* __restrict__ W0ch,
                const _Float16* __restrict__ W1h, const _Float16* __restrict__ W2h,
                const _Float16* __restrict__ WHh,
                float* __restrict__ outp, int B) {
  extern __shared__ char smem[];
  _Float16* Tin  = (_Float16*)(smem + SMEM_TIN);
  _Float16* Tout = (_Float16*)(smem + SMEM_TOUT);
  _Float16* hA   = (_Float16*)(smem + SMEM_TOUT);            // alias: forward only
  _Float16* hB   = (_Float16*)(smem + SMEM_TOUT + 16 * HLD * 2);
  unsigned* mk0  = (unsigned*)(smem + SMEM_MISC);
  unsigned* mk1  = mk0 + 256;
  unsigned* mk2  = mk1 + 256;
  float* diag_s  = (float*)(mk2 + 256);
  float* tril_s  = diag_s + 16 * 7;
  float* Jd      = tril_s + 16 * 21;
  float* Jt      = Jd + 16 * 49;

  const int tid  = threadIdx.x;
  const int lane = tid & 31;
  const int wave = tid >> 5;
  const int base = blockIdx.x * 16;
  const int colL = lane & 15;
  const int rowB = (lane >> 4) << 3;
  const size_t Bl = (size_t)B;
  if (base >= B) return;

  for (int i = tid; i < 768; i += 256) {
    if (i < 256) mk0[i] = 0u;
    else if (i < 512) mk1[i - 256] = 0u;
    else mk2[i - 512] = 0u;
  }
  __syncthreads();

  // ---- layer 0: q(16x7 pad32) @ W0^T ----
  {
    v8f z = {0, 0, 0, 0, 0, 0, 0, 0};
    v16h a = ldA(qh + (size_t)base * 32, 32, lane);
#pragma unroll
    for (int u = 0; u < 4; ++u) {
      const int nt = wave * 4 + u;
      v16h b = ldBv(W0h + (size_t)nt * 16 * 32, 32, lane);
      v8f acc = (u == 0) ? wmma_f16(a, b, z) : wmma_f16_ra(a, b, z);
      const int n = nt * 16 + colL;
      const float bias = b0[n];
#pragma unroll
      for (int j = 0; j < 8; ++j) {
        const int sl = rowB + j;
        float pre = acc[j] + bias;
        hA[sl * HLD + n] = (_Float16)(pre > 0.f ? pre : 0.f);
        atomicOr(&mk0[sl * 16 + (n >> 5)], pre > 0.f ? (1u << (n & 31)) : 0u);
      }
    }
  }
  __syncthreads();

  // ---- layers 1, 2 ----
  fwd_layer(hA, W1h, b1, hB, mk1, lane, wave);
  __syncthreads();
  fwd_layer(hB, W2h, b2, hA, mk2, lane, wave);   // h2 in hA
  __syncthreads();

  // ---- forward heads: diag(relu) + tril ----
  if (wave < 2) {
    const int nt = wave;
    v8f acc = {0, 0, 0, 0, 0, 0, 0, 0};
    for (int k0 = 0; k0 < HIDN; k0 += 32) {
      v16h a = ldA(hA + k0, HLD, lane);
      v16h b = ldBv(WHh + (size_t)nt * 16 * HIDN + k0, HIDN, lane);
      acc = wmma_f16(a, b, acc);
    }
    const int n = nt * 16 + colL;
#pragma unroll
    for (int j = 0; j < 8; ++j) {
      const int sl = rowB + j;
      const size_t s = (size_t)base + sl;
      float v = acc[j];
      if (n < 7) {
        v += bd[n];
        v = v > 0.f ? v : 0.f;
        diag_s[sl * 7 + n] = v;
        outp[s * 7 + n] = v;                               // out_L_diag
      } else if (n < 28) {
        v += bt[n - 7];
        tril_s[sl * 21 + (n - 7)] = v;
        outp[7 * Bl + s * 21 + (n - 7)] = v;               // out_L_tril
      }
    }
  }
  __syncthreads();

  // ---- build T0 (all 8 pairs, 128 cols) : relu'(pre0) .* W0, vectorized ----
  for (int idx = tid; idx < 128 * 64; idx += 256) {
    const int col = idx >> 6;
    const int k8  = (idx & 63) << 3;
    const int c   = col & 15;
    const int g   = col >> 4;
    v8h tv;
    if (c < 14) {
      const int s = g * 2 + ((c < 7) ? 0 : 1);
      const int j = (c < 7) ? c : c - 7;
      const unsigned mw = mk0[s * 16 + (k8 >> 5)] >> (k8 & 31);
      v8h w = *(const v8h*)(W0ch + (size_t)j * HIDN + k8);
#pragma unroll
      for (int i = 0; i < 8; ++i)
        tv[i] = ((mw >> i) & 1u) ? w[i] : (_Float16)0.f;
    } else {
#pragma unroll
      for (int i = 0; i < 8; ++i) tv[i] = (_Float16)0.f;
    }
    *(v8h*)(Tin + (size_t)col * LDP + k8) = tv;
  }
  __syncthreads();

  // ---- tangent propagation through layers 1 and 2 (batched over 8 pairs) ----
  jac_level(W1h, Tin, mk1, Tout, lane, wave);    // T1
  __syncthreads();
  jac_level(W2h, Tout, mk2, Tin, lane, wave);    // T2 (back into Tin)
  __syncthreads();

  // ---- head Jacobians: 16 (mt,g) combos spread over 8 waves ----
  {
#pragma unroll
    for (int t2 = 0; t2 < 2; ++t2) {
      const int combo = wave * 2 + t2;     // 0..15
      const int mt = combo >> 3;           // 0..1
      const int g  = combo & 7;            // pair
      v8f acc = {0, 0, 0, 0, 0, 0, 0, 0};
      for (int k0 = 0; k0 < HIDN; k0 += 32) {
        v16h a = ldA(WHh + (size_t)mt * 16 * HIDN + k0, HIDN, lane);
        v16h b = ldBv(Tin + (size_t)(g * 16) * LDP + k0, LDP, lane);
        acc = wmma_f16(a, b, acc);
      }
      const int c = colL;
      if (c < 14) {
        const int si = (c < 7) ? 0 : 1;
        const int jc = (c < 7) ? c : c - 7;
        const int sl = g * 2 + si;
#pragma unroll
        for (int j = 0; j < 8; ++j) {
          const int r = mt * 16 + rowB + j;
          float v = acc[j];
          if (r < 7) {
            v = (diag_s[sl * 7 + r] > 0.f) ? v : 0.f;
            Jd[sl * 49 + r * 7 + jc] = v;
          } else if (r < 28) {
            Jt[sl * 147 + (r - 7) * 7 + jc] = v;
          }
        }
      }
    }
  }
  __syncthreads();

  // ---- assembly: L, L_dq, transposes, H, L_dt ----
  {
    const int g = tid >> 4;        // local sample
    const int t = tid & 15;
    const size_t s = (size_t)base + g;
    float* oddq  = outp + 28 * Bl;
    float* otdq  = outp + 77 * Bl;
    float* oL    = outp + 224 * Bl;
    float* oLdq  = outp + 273 * Bl;
    float* oLT   = outp + 616 * Bl;
    float* oLdqT = outp + 665 * Bl;
    float* oH    = outp + 1008 * Bl;
    float* oLdt  = outp + 1057 * Bl;
    float* oLdtT = outp + 1106 * Bl;

    for (int e = t; e < 49; e += 16)  oddq[s * 49 + e]  = Jd[g * 49 + e];
    for (int e = t; e < 147; e += 16) otdq[s * 147 + e] = Jt[g * 147 + e];

    float qdv[7];
#pragma unroll
    for (int k = 0; k < 7; ++k) qdv[k] = qd[s * 7 + k];

    for (int e = t; e < 49; e += 16) {
      int i = e / 7, j = e % 7;
      oL[s * 49 + e]  = Lget(diag_s, tril_s, g, i, j);
      oLT[s * 49 + e] = Lget(diag_s, tril_s, g, j, i);
      float hv = 0.f, dt = 0.f, dtT = 0.f;
#pragma unroll
      for (int k = 0; k < 7; ++k) {
        hv  += Lget(diag_s, tril_s, g, i, k) * Lget(diag_s, tril_s, g, j, k);
        dt  += LdqGet(Jd, Jt, g, i, j, k) * qdv[k];
        dtT += LdqGet(Jd, Jt, g, j, i, k) * qdv[k];
      }
      oH[s * 49 + e]    = hv;
      oLdt[s * 49 + e]  = dt;
      oLdtT[s * 49 + e] = dtT;
    }
    for (int e = t; e < 343; e += 16) {
      int i = e / 49, j = (e / 7) % 7, k = e % 7;
      oLdq[s * 343 + e]  = LdqGet(Jd, Jt, g, i, j, k);
      oLdqT[s * 343 + e] = LdqGet(Jd, Jt, g, j, i, k);
    }
  }
}

// ---------------- launch ----------------

extern "C" void kernel_launch(void* const* d_in, const int* in_sizes, int n_in,
                              void* d_out, int out_size, void* d_ws, size_t ws_size,
                              hipStream_t stream) {
  (void)n_in; (void)out_size; (void)ws_size;
  const float* q  = (const float*)d_in[0];
  const float* qd = (const float*)d_in[1];
  const float* W0 = (const float*)d_in[3];
  const float* b0 = (const float*)d_in[4];
  const float* W1 = (const float*)d_in[5];
  const float* b1 = (const float*)d_in[6];
  const float* W2 = (const float*)d_in[7];
  const float* b2 = (const float*)d_in[8];
  const float* Wd = (const float*)d_in[9];
  const float* bd = (const float*)d_in[10];
  const float* Wt = (const float*)d_in[11];
  const float* bt = (const float*)d_in[12];
  float* outp = (float*)d_out;
  const int B = in_sizes[0] / NDOF;

  _Float16* qh   = (_Float16*)d_ws;
  _Float16* W0h  = qh + (size_t)B * 32;
  _Float16* W0ch = W0h + HIDN * 32;
  _Float16* W1h  = W0ch + 8 * HIDN;
  _Float16* W2h  = W1h + HIDN * HIDN;
  _Float16* WHh  = W2h + HIDN * HIDN;

  k_prep_q<<<(B * 32 + 255) / 256, 256, 0, stream>>>(q, qh, B);
  k_prep_w0<<<(HIDN * 32 + 255) / 256, 256, 0, stream>>>(W0, W0h);
  k_prep_w0c<<<(8 * HIDN + 255) / 256, 256, 0, stream>>>(W0, W0ch);
  k_cvt<<<(HIDN * HIDN + 255) / 256, 256, 0, stream>>>(W1, W1h, HIDN * HIDN);
  k_cvt<<<(HIDN * HIDN + 255) / 256, 256, 0, stream>>>(W2, W2h, HIDN * HIDN);
  k_prep_heads<<<(32 * HIDN + 255) / 256, 256, 0, stream>>>(Wd, Wt, WHh);

  (void)hipFuncSetAttribute(reinterpret_cast<const void*>(delan_mega),
                            hipFuncAttributeMaxDynamicSharedMemorySize, SMEM_BYTES);
  delan_mega<<<B / 16, 256, SMEM_BYTES, stream>>>(qd, b0, b1, b2, bd, bt,
                                                  qh, W0h, W0ch, W1h, W2h, WHh,
                                                  outp, B);
}
